// MultiHeadAttentionBlock_77318001262899
// MI455X (gfx1250) — compile-verified, matrix operand order checked by reading, not run on hardware
//
#include <hip/hip_runtime.h>
#include <hip/hip_bf16.h>

typedef __bf16 bf16;
typedef __attribute__((ext_vector_type(16))) __bf16 v16bf;
typedef __attribute__((ext_vector_type(8)))  float  v8f;

#define B_ 4
#define N_ 2048
#define E_ 512
#define H_ 8
#define D_ 64

// ---------------------------------------------------------------------------
// Fragment loader for 16x32 bf16 A/B tiles (wave32 layout per CDNA5 ISA §7.12.2):
//   lane L in [0,15]:  row M=L,     holds K = k0+0..7   and K = k0+16..23
//   lane L in [16,31]: row M=L-16,  holds K = k0+8..15  and K = k0+24..31
// tile_base points at row 0 of the tile; ld = row stride in elements.
// ---------------------------------------------------------------------------
static __device__ inline v16bf load_frag(const bf16* tile_base, int ld, int k0) {
    const int lane = threadIdx.x & 31;
    const bf16* p = tile_base + (size_t)(lane & 15) * ld + (k0 + ((lane >> 4) & 1) * 8);
    union { v16bf v; uint4 u[2]; } f;
    f.u[0] = *(const uint4*)(p);        // elements 0..7   (K = k0 + kh*8 + 0..7)
    f.u[1] = *(const uint4*)(p + 16);   // elements 8..15  (K = k0 + 16 + kh*8 + 0..7)
    return f.v;
}

static __device__ inline v8f wmma_bf16(v16bf a, v16bf b, v8f c) {
    return __builtin_amdgcn_wmma_f32_16x16x32_bf16(
        /*neg_a=*/false, a, /*neg_b=*/false, b,
        /*c_mod=*/(short)0, c, /*reuse_a=*/false, /*reuse_b=*/false);
}

// ---------------------------------------------------------------------------
// Prep kernels: fp32 -> bf16 convert, and weight transpose to [N,K] so that
// B-matrix fragments are contiguous per-lane 16B loads.
// ---------------------------------------------------------------------------
__global__ __launch_bounds__(256) void cvt_f32_bf16(const float* __restrict__ in,
                                                    bf16* __restrict__ out, int n) {
    int i = blockIdx.x * 256 + threadIdx.x;
    if (i < n) out[i] = (bf16)in[i];
}

__global__ __launch_bounds__(256) void transpose_w(const float* __restrict__ w,
                                                   bf16* __restrict__ wt,
                                                   int K, int Ncol) {
    // w: [K, Ncol] fp32 -> wt: [Ncol, K] bf16
    int i = blockIdx.x * 256 + threadIdx.x;
    if (i < K * Ncol) {
        int kk = i / Ncol, nn = i % Ncol;
        wt[(size_t)nn * K + kk] = (bf16)w[i];
    }
}

// ---------------------------------------------------------------------------
// Kernel 1: QKV GEMM.  [8192,1536] = x_bf16[8192,512] @ WqkvT[1536,512]^T + b
// Each wave computes a 32x64 output tile: 8 accumulators, 2 A-fragments and
// 4 B-fragments per k-step -> 8 WMMAs at 1.5 b128 loads per WMMA.
// Output scattered into q [B,H,N,D], k [B,H,N,D], vT [B,H,D,N] (bf16).
// ---------------------------------------------------------------------------
__global__ __launch_bounds__(256) void qkv_gemm(const bf16* __restrict__ xb,
                                                const bf16* __restrict__ wT,
                                                const float* __restrict__ bias,
                                                bf16* __restrict__ q,
                                                bf16* __restrict__ k,
                                                bf16* __restrict__ vT) {
    const int NQ = (3 * E_) / 64;                 // 24 quad-column tiles
    int tile = blockIdx.x * 8 + (threadIdx.x >> 5);
    int mt = tile / NQ, nq = tile % NQ;
    int m0 = mt * 32, n0 = nq * 64;

    const bf16* am0 = xb + (size_t)(m0 +  0) * E_;
    const bf16* am1 = xb + (size_t)(m0 + 16) * E_;
    const bf16* b0 = wT + (size_t)(n0 +  0) * E_;
    const bf16* b1 = wT + (size_t)(n0 + 16) * E_;
    const bf16* b2 = wT + (size_t)(n0 + 32) * E_;
    const bf16* b3 = wT + (size_t)(n0 + 48) * E_;

    v8f c00 = {}, c01 = {}, c02 = {}, c03 = {};
    v8f c10 = {}, c11 = {}, c12 = {}, c13 = {};
#pragma unroll 2
    for (int k0 = 0; k0 < E_; k0 += 32) {
        v16bf af0 = load_frag(am0, E_, k0);
        v16bf af1 = load_frag(am1, E_, k0);
        v16bf bf0 = load_frag(b0, E_, k0);
        v16bf bf1 = load_frag(b1, E_, k0);
        v16bf bf2 = load_frag(b2, E_, k0);
        v16bf bf3 = load_frag(b3, E_, k0);
        c00 = wmma_bf16(af0, bf0, c00);
        c01 = wmma_bf16(af0, bf1, c01);
        c02 = wmma_bf16(af0, bf2, c02);
        c03 = wmma_bf16(af0, bf3, c03);
        c10 = wmma_bf16(af1, bf0, c10);
        c11 = wmma_bf16(af1, bf1, c11);
        c12 = wmma_bf16(af1, bf2, c12);
        c13 = wmma_bf16(af1, bf3, c13);
    }

    const int lane = threadIdx.x & 31;
    const int kh   = (lane >> 4) & 1;
    v8f accs[2][4] = {{c00, c01, c02, c03}, {c10, c11, c12, c13}};

#pragma unroll
    for (int j = 0; j < 4; ++j) {
        const int col = n0 + j * 16 + (lane & 15);  // column in [0,1536)
        const int qkv = col / E_;                   // 0=q,1=k,2=v
        const int h   = (col % E_) / D_;
        const int d   = col % D_;
        const float bv = bias[col];
#pragma unroll
        for (int mi = 0; mi < 2; ++mi) {
#pragma unroll
            for (int i = 0; i < 8; ++i) {
                int row = m0 + mi * 16 + i + 8 * kh;    // in [0, B*N)
                int b   = row / N_;
                int n   = row % N_;
                bf16 val = (bf16)(accs[mi][j][i] + bv);
                if (qkv == 0)      q [((size_t)(b * H_ + h) * N_ + n) * D_ + d] = val;
                else if (qkv == 1) k [((size_t)(b * H_ + h) * N_ + n) * D_ + d] = val;
                else               vT[((size_t)(b * H_ + h) * D_ + d) * N_ + n] = val;
            }
        }
    }
}

// ---------------------------------------------------------------------------
// Kernel 2: attention with softmax over the HEAD axis (faithful to reference).
// One block per (b, 16-query tile); wave w handles head h=w.
// k-loop in tiles of 32: S via WMMA -> LDS, cross-head softmax / sqrt(E),
// bf16 P -> LDS, then P@V via WMMA accumulated in registers.
// ---------------------------------------------------------------------------
__global__ __launch_bounds__(256) void attention(const bf16* __restrict__ q,
                                                 const bf16* __restrict__ kmat,
                                                 const bf16* __restrict__ vT,
                                                 bf16* __restrict__ out) {
    __shared__ float sS[H_][16][32];   // 16 KB: scores per head
    __shared__ bf16  sP[H_][16][32];   //  8 KB: normalized probs (bf16)

    const int b  = blockIdx.x / (N_ / 16);
    const int qt = blockIdx.x % (N_ / 16);
    const int q0 = qt * 16;
    const int h    = threadIdx.x >> 5;            // wave == head
    const int lane = threadIdx.x & 31;
    const int kh   = (lane >> 4) & 1;

    const bf16* qbase = q    + ((size_t)(b * H_ + h) * N_ + q0) * D_;
    const bf16* kbase = kmat +  (size_t)(b * H_ + h) * N_ * D_;
    const bf16* vbase = vT   +  (size_t)(b * H_ + h) * D_ * N_;

    // Q fragments for d=0..31 and d=32..63, kept resident
    const v16bf qf0 = load_frag(qbase, D_, 0);
    const v16bf qf1 = load_frag(qbase, D_, 32);

    v8f o0 = {}, o1 = {}, o2 = {}, o3 = {};
    const float inv_scale = 0.044194173824159216f;   // 1/sqrt(512)

    for (int kt = 0; kt < N_; kt += 32) {
        // ---- S = Q @ K^T for this head's 16x32 slab (two 16x16 C tiles) ----
#pragma unroll
        for (int half = 0; half < 2; ++half) {
            const bf16* krows = kbase + (size_t)(kt + half * 16) * D_;
            v8f s = {};
            s = wmma_bf16(qf0, load_frag(krows, D_, 0),  s);
            s = wmma_bf16(qf1, load_frag(krows, D_, 32), s);
#pragma unroll
            for (int i = 0; i < 8; ++i)
                sS[h][i + 8 * kh][half * 16 + (lane & 15)] = s[i];
        }

        // prefetch next slab's K rows and V row segments (global_prefetch_b8)
        if (kt + 32 < N_) {
            __builtin_prefetch(kbase + (size_t)(kt + 32) * D_ + (size_t)lane * 64, 0, 0);
            __builtin_prefetch(vbase + (size_t)(lane * 2) * N_ + (kt + 32), 0, 0);
        }
        __syncthreads();

        // ---- softmax over heads at each (q,k) position; then /sqrt(E) ----
#pragma unroll
        for (int p = threadIdx.x; p < 512; p += 256) {
            int qi = p >> 5, ki = p & 31;
            float m = sS[0][qi][ki];
#pragma unroll
            for (int hh = 1; hh < H_; ++hh) m = fmaxf(m, sS[hh][qi][ki]);
            float e[H_]; float sum = 0.f;
#pragma unroll
            for (int hh = 0; hh < H_; ++hh) { e[hh] = __expf(sS[hh][qi][ki] - m); sum += e[hh]; }
            float r = inv_scale / sum;
#pragma unroll
            for (int hh = 0; hh < H_; ++hh) sP[hh][qi][ki] = (bf16)(e[hh] * r);
        }
        __syncthreads();

        // ---- O += P @ V  (A = P 16x32 from LDS, B = V^T rows contiguous) ----
        v16bf pf = load_frag(&sP[h][0][0], 32, 0);
        o0 = wmma_bf16(pf, load_frag(vbase + (size_t)0  * N_, N_, kt), o0);
        o1 = wmma_bf16(pf, load_frag(vbase + (size_t)16 * N_, N_, kt), o1);
        o2 = wmma_bf16(pf, load_frag(vbase + (size_t)32 * N_, N_, kt), o2);
        o3 = wmma_bf16(pf, load_frag(vbase + (size_t)48 * N_, N_, kt), o3);
        __syncthreads();   // protect LDS before next iteration overwrites
    }

    // ---- store: out[b, n, h*64 + d]  (layout 'b n (h d)') ----
    v8f oa[4] = {o0, o1, o2, o3};
#pragma unroll
    for (int j = 0; j < 4; ++j) {
#pragma unroll
        for (int i = 0; i < 8; ++i) {
            int n = q0 + i + 8 * kh;
            int e = h * D_ + j * 16 + (lane & 15);
            out[((size_t)b * N_ + n) * E_ + e] = (bf16)oa[j][i];
        }
    }
}

// ---------------------------------------------------------------------------
// Kernel 3: output projection. d_out[8192,512] = ao @ WprojT^T + bproj (fp32)
// 32x64 tile per wave, same scheme as kernel 1.
// ---------------------------------------------------------------------------
__global__ __launch_bounds__(256) void proj_gemm(const bf16* __restrict__ a,
                                                 const bf16* __restrict__ wT,
                                                 const float* __restrict__ bias,
                                                 float* __restrict__ out) {
    const int NQ = E_ / 64;                       // 8 quad-column tiles
    int tile = blockIdx.x * 8 + (threadIdx.x >> 5);
    int mt = tile / NQ, nq = tile % NQ;
    int m0 = mt * 32, n0 = nq * 64;

    const bf16* am0 = a + (size_t)(m0 +  0) * E_;
    const bf16* am1 = a + (size_t)(m0 + 16) * E_;
    const bf16* b0 = wT + (size_t)(n0 +  0) * E_;
    const bf16* b1 = wT + (size_t)(n0 + 16) * E_;
    const bf16* b2 = wT + (size_t)(n0 + 32) * E_;
    const bf16* b3 = wT + (size_t)(n0 + 48) * E_;

    v8f c00 = {}, c01 = {}, c02 = {}, c03 = {};
    v8f c10 = {}, c11 = {}, c12 = {}, c13 = {};
#pragma unroll 2
    for (int k0 = 0; k0 < E_; k0 += 32) {
        v16bf af0 = load_frag(am0, E_, k0);
        v16bf af1 = load_frag(am1, E_, k0);
        v16bf bf0 = load_frag(b0, E_, k0);
        v16bf bf1 = load_frag(b1, E_, k0);
        v16bf bf2 = load_frag(b2, E_, k0);
        v16bf bf3 = load_frag(b3, E_, k0);
        c00 = wmma_bf16(af0, bf0, c00);
        c01 = wmma_bf16(af0, bf1, c01);
        c02 = wmma_bf16(af0, bf2, c02);
        c03 = wmma_bf16(af0, bf3, c03);
        c10 = wmma_bf16(af1, bf0, c10);
        c11 = wmma_bf16(af1, bf1, c11);
        c12 = wmma_bf16(af1, bf2, c12);
        c13 = wmma_bf16(af1, bf3, c13);
    }

    const int lane = threadIdx.x & 31;
    const int kh   = (lane >> 4) & 1;
    v8f accs[2][4] = {{c00, c01, c02, c03}, {c10, c11, c12, c13}};
#pragma unroll
    for (int j = 0; j < 4; ++j) {
        const int col = n0 + j * 16 + (lane & 15);
        const float bv = bias[col];
#pragma unroll
        for (int mi = 0; mi < 2; ++mi) {
#pragma unroll
            for (int i = 0; i < 8; ++i) {
                int row = m0 + mi * 16 + i + 8 * kh;
                out[(size_t)row * E_ + col] = accs[mi][j][i] + bv;
            }
        }
    }
}

// ---------------------------------------------------------------------------
extern "C" void kernel_launch(void* const* d_in, const int* in_sizes, int n_in,
                              void* d_out, int out_size, void* d_ws, size_t ws_size,
                              hipStream_t stream) {
    const float* x     = (const float*)d_in[0];   // [4,2048,512]
    const float* Wqkv  = (const float*)d_in[1];   // [512,1536]
    const float* bqkv  = (const float*)d_in[2];   // [1536]
    const float* Wproj = (const float*)d_in[3];   // [512,512]
    const float* bproj = (const float*)d_in[4];   // [512]
    float* out = (float*)d_out;                   // [4,2048,512] fp32

    // bf16 workspace carve-up (~42 MB total)
    bf16* ws = (bf16*)d_ws;
    size_t o = 0;
    bf16* xb     = ws + o; o += (size_t)B_ * N_ * E_;        // 4M
    bf16* wqkvT  = ws + o; o += (size_t)3 * E_ * E_;         // 768K
    bf16* wprojT = ws + o; o += (size_t)E_ * E_;             // 256K
    bf16* qb     = ws + o; o += (size_t)B_ * H_ * N_ * D_;   // 4M
    bf16* kb     = ws + o; o += (size_t)B_ * H_ * N_ * D_;   // 4M
    bf16* vtb    = ws + o; o += (size_t)B_ * H_ * N_ * D_;   // 4M  (V transposed: [B,H,D,N])
    bf16* ao     = ws + o; o += (size_t)B_ * N_ * E_;        // 4M  (attn out, 'b n (h d)')

    // prep: converts + weight transposes
    {
        int n = B_ * N_ * E_;
        cvt_f32_bf16<<<(n + 255) / 256, 256, 0, stream>>>(x, xb, n);
    }
    transpose_w<<<(E_ * 3 * E_ + 255) / 256, 256, 0, stream>>>(Wqkv, wqkvT, E_, 3 * E_);
    transpose_w<<<(E_ * E_ + 255) / 256, 256, 0, stream>>>(Wproj, wprojT, E_, E_);

    // QKV: 256 row-tiles (32 rows) x 24 quad-col tiles = 6144 wave-tiles / 8
    qkv_gemm<<<768, 256, 0, stream>>>(xb, wqkvT, bqkv, qb, kb, vtb);

    // attention: one block per (b, 16-query tile) = 4 * 128 = 512 blocks
    attention<<<512, 256, 0, stream>>>(qb, kb, vtb, ao);

    // projection: 256 x 8 = 2048 wave-tiles / 8 per block
    proj_gemm<<<256, 256, 0, stream>>>(ao, wprojT, bproj, out);
}
